// Physics_7370163880185
// MI455X (gfx1250) — matrix operand order repaired and checked by reading, not compiled, not run on hardware
//
#include <hip/hip_runtime.h>

// MI455X / gfx1250, wave32. Strategy: bf16 WMMA (v_wmma_f32_16x16x32_bf16),
// factor pair-MLP layer 1 out of the NxN pair loop, fuse everything so HBM
// traffic is ~60MB and the kernel is WMMA-throughput bound (~110 GFLOP).
// v3: force 4x unroll of the K loops -> 8 WMMAs per loop body across two
// independent accumulator chains (dense matrix-unit issue, LDS loads
// software-pipelined by the scheduler).

typedef __attribute__((ext_vector_type(16))) __bf16 v16bf;
typedef __attribute__((ext_vector_type(8)))  float  v8f;
typedef unsigned int   u32;
typedef unsigned short u16;

__device__ __forceinline__ u16 f2bf(float f) {
  u32 u = __float_as_uint(f);
  u32 r = u + 0x7fffu + ((u >> 16) & 1u);   // round-to-nearest-even
  return (u16)(r >> 16);
}
__device__ __forceinline__ u32 pack2(float lo, float hi) {
  return (u32)f2bf(lo) | ((u32)f2bf(hi) << 16);
}

union FragAB { v16bf v; u32 u[8]; };

// A-matrix 16x32 bf16 fragment from row-major LDS tile (ISA 7.12.2):
// lanes 0-15: M=lane, VGPR v -> K = 16*(v>>2) + 2*(v&3)   (+8 for lanes 16-31)
__device__ __forceinline__ v16bf load_fragA(const u16* base, int stride, int lane) {
  FragAB f;
  const int row = lane & 15, half = lane >> 4;
  const u16* p = base + row * stride + 8 * half;
#pragma unroll
  for (int v = 0; v < 8; ++v) {
    const int k = 16 * (v >> 2) + 2 * (v & 3);
    f.u[v] = *(const u32*)(p + k);
  }
  return f.v;
}

// B-matrix 32x16 bf16 fragment; tile stored TRANSPOSED in LDS: (k,n) at base[n*stride+k].
// lanes 0-15: N=lane, VGPR v -> K = 16*(v>>2) + 2*(v&3)   (+8 for lanes 16-31)
__device__ __forceinline__ v16bf load_fragB(const u16* base, int stride, int lane) {
  FragAB f;
  const int col = lane & 15, half = lane >> 4;
  const u16* p = base + col * stride + 8 * half;
#pragma unroll
  for (int v = 0; v < 8; ++v) f.u[v] = *(const u32*)(p + 16 * (v >> 2) + 2 * (v & 3));
  return f.v;
}

__device__ __forceinline__ v8f wmma_bf16(v16bf a, v16bf b, v8f c) {
  return __builtin_amdgcn_wmma_f32_16x16x32_bf16(false, a, false, b, (short)0, c, false, false);
}

__global__ void f32_to_bf16(const float* __restrict__ src, u16* __restrict__ dst, int n) {
  int i = blockIdx.x * blockDim.x + threadIdx.x;
  if (i * 2 + 1 < n) {
    float2 v = *(const float2*)(src + (size_t)i * 2);
    *(u32*)(dst + (size_t)i * 2) = pack2(v.x, v.y);
  }
}

// Generic C = act(A[MxK] @ W[KxN] + b), A/W bf16 row-major.
// mode 0: bf16 out, relu            (hidden layers)
// mode 1: f32 out, raw              (P1/P2: bias+relu deferred to pair kernel)
// mode 2: f32 out, relu + residual  (final transition layer -> d_out)
// Block: 256 thr (8 waves), tile 32x64, one 16x16 WMMA tile per wave.
__global__ __launch_bounds__(256) void gemm_bias_act(
    const u16* __restrict__ A, const u16* __restrict__ W,
    const float* __restrict__ bias, void* __restrict__ out,
    const float* __restrict__ residual, int M, int K, int N, int mode)
{
  __shared__ u16 Asm[32 * 40];   // 32 rows x 32 K, stride 40 (pad vs bank conflicts)
  __shared__ u16 Bsm[64 * 40];   // 64 n  x 32 K, transposed panel

  const int m0 = blockIdx.x * 32;
  const int n0 = blockIdx.y * 64;
  const int tid  = threadIdx.x;
  const int lane = tid & 31;
  const int wave = tid >> 5;
  const int tm = wave >> 2, tn = wave & 3;

  v8f acc = {};

  for (int k0 = 0; k0 < K; k0 += 32) {
    __syncthreads();
#pragma unroll
    for (int s = 0; s < 2; ++s) {                 // A tile 32x32
      int idx = tid + s * 256;
      int row = idx >> 4, kp = (idx & 15) * 2;
      u32 v = *(const u32*)(A + (size_t)(m0 + row) * K + k0 + kp);
      *(u32*)(&Asm[row * 40 + kp]) = v;
    }
#pragma unroll
    for (int s = 0; s < 4; ++s) {                 // W panel 32x64, transpose in
      int idx = tid + s * 256;
      int k = idx >> 5, n = (idx & 31) * 2;
      u32 v = *(const u32*)(W + (size_t)(k0 + k) * N + n0 + n);
      Bsm[n * 40 + k]       = (u16)v;
      Bsm[(n + 1) * 40 + k] = (u16)(v >> 16);
    }
    __syncthreads();
    v16bf a = load_fragA(&Asm[tm * 16 * 40], 40, lane);
    v16bf b = load_fragB(&Bsm[tn * 16 * 40], 40, lane);
    acc = wmma_bf16(a, b, acc);
  }

  const int col = lane & 15, half = lane >> 4;
  const int gcol  = n0 + tn * 16 + col;
  const int growb = m0 + tm * 16 + 8 * half;      // C/D: M = v + 8*half, N = col
  if (mode == 0) {
    u16* dst = (u16*)out;
    float bv = bias[gcol];
#pragma unroll
    for (int v = 0; v < 8; ++v) {
      float x = acc[v] + bv; x = x > 0.f ? x : 0.f;
      dst[(size_t)(growb + v) * N + gcol] = f2bf(x);
    }
  } else if (mode == 1) {
    float* dst = (float*)out;
#pragma unroll
    for (int v = 0; v < 8; ++v)
      dst[(size_t)(growb + v) * N + gcol] = acc[v];
  } else {
    float* dst = (float*)out;
    float bv = bias[gcol];
#pragma unroll
    for (int v = 0; v < 8; ++v) {
      size_t o = (size_t)(growb + v) * N + gcol;
      float x = acc[v] + bv; x = x > 0.f ? x : 0.f;
      dst[o] = x + residual[o];                   // out = relu(trans3) + objs
    }
  }
}

// Pair kernel: block = (row-block rb, j, i), 256 thr (8 waves), 64 batch rows.
// h1 = relu(P1[i]+P2[j]+b1) -> LDS bf16; h2 = relu(h1@W2+b2) -> LDS bf16;
// h3 = relu(h2@W3+b3) -> atomicAdd into out[i]. Dynamic LDS = 3*64*520*2 B.
#define H1S 520
__global__ __launch_bounds__(256) void pair_kernel(
    const float* __restrict__ P1, const float* __restrict__ P2,
    const float* __restrict__ ib1, const u16* __restrict__ W2,
    const float* __restrict__ ib2, const u16* __restrict__ W3,
    const float* __restrict__ ib3, float* __restrict__ out)
{
  const int rb = blockIdx.x;   // 0..7  (64-row blocks of B=512)
  const int j  = blockIdx.y;   // 0..15
  const int i  = blockIdx.z;   // 0..15
  if (i == j) return;          // uniform: EXEC stays all-1s for WMMA blocks

  extern __shared__ char smem[];
  u16* h1 = (u16*)smem;        // [64][520]
  u16* h2 = h1 + 64 * H1S;     // [64][520]
  u16* wt = h2 + 64 * H1S;     // [64 n][520] transposed weight panel

  const int tid  = threadIdx.x;
  const int lane = tid & 31;
  const int wave = tid >> 5;
  const int col  = lane & 15, half = lane >> 4;
  // each wave owns two adjacent-N tiles of the same 16-row band:
  const int tm  = (wave * 2) >> 2;          // 0..3
  const int tn0 = (wave * 2) & 3;           // 0 or 2
  const int tn1 = tn0 + 1;

  // ---- stage 1: h1 = relu(P1[i,rows] + P2[j,rows] + b1), fp32 add -> bf16
  const size_t ri = (size_t)i * 512 + rb * 64;
  const size_t rj = (size_t)j * 512 + rb * 64;
#pragma unroll 4
  for (int it = 0; it < 32; ++it) {
    int e = (tid + it * 256) * 4;
    int r = e >> 9, c = e & 511;
    float4 p1 = *(const float4*)(P1 + (ri + r) * 512 + c);
    float4 p2 = *(const float4*)(P2 + (rj + r) * 512 + c);
    float4 bb = *(const float4*)(ib1 + c);
    float x0 = p1.x + p2.x + bb.x; x0 = x0 > 0.f ? x0 : 0.f;
    float x1 = p1.y + p2.y + bb.y; x1 = x1 > 0.f ? x1 : 0.f;
    float x2 = p1.z + p2.z + bb.z; x2 = x2 > 0.f ? x2 : 0.f;
    float x3 = p1.w + p2.w + bb.w; x3 = x3 > 0.f ? x3 : 0.f;
    *(u32*)(&h1[r * H1S + c])     = pack2(x0, x1);
    *(u32*)(&h1[r * H1S + c + 2]) = pack2(x2, x3);
  }
  __syncthreads();

  // ---- stage 2: h2 = relu(h1 @ W2 + b2), 8 column chunks of 64
  for (int nc = 0; nc < 8; ++nc) {
    for (int s = 0; s < 64; ++s) {               // W2[:,nc*64..+64] transposed
      int idx = tid + s * 256;
      int k = idx >> 5, n = (idx & 31) * 2;
      u32 v = *(const u32*)(W2 + (size_t)k * 512 + nc * 64 + n);
      wt[n * H1S + k]       = (u16)v;
      wt[(n + 1) * H1S + k] = (u16)(v >> 16);
    }
    __syncthreads();
    v8f acc0 = {}, acc1 = {};
#pragma unroll 4
    for (int kt = 0; kt < 16; ++kt) {
      v16bf a  = load_fragA(&h1[tm * 16 * H1S + kt * 32], H1S, lane);   // shared
      v16bf b0 = load_fragB(&wt[tn0 * 16 * H1S + kt * 32], H1S, lane);
      v16bf b1 = load_fragB(&wt[tn1 * 16 * H1S + kt * 32], H1S, lane);
      acc0 = wmma_bf16(a, b0, acc0);
      acc1 = wmma_bf16(a, b1, acc1);
    }
    float bv0 = ib2[nc * 64 + tn0 * 16 + col];
    float bv1 = ib2[nc * 64 + tn1 * 16 + col];
#pragma unroll
    for (int v = 0; v < 8; ++v) {
      int row = tm * 16 + v + 8 * half;
      float x0 = acc0[v] + bv0; x0 = x0 > 0.f ? x0 : 0.f;
      float x1 = acc1[v] + bv1; x1 = x1 > 0.f ? x1 : 0.f;
      h2[row * H1S + nc * 64 + tn0 * 16 + col] = f2bf(x0);
      h2[row * H1S + nc * 64 + tn1 * 16 + col] = f2bf(x1);
    }
    __syncthreads();
  }

  // ---- stage 3: h3 = relu(h2 @ W3 + b3) -> atomicAdd out[i] (sum over j)
  for (int nc = 0; nc < 4; ++nc) {
    for (int s = 0; s < 64; ++s) {               // W3[:,nc*64..+64] transposed
      int idx = tid + s * 256;
      int k = idx >> 5, n = (idx & 31) * 2;
      u32 v = *(const u32*)(W3 + (size_t)k * 256 + nc * 64 + n);
      wt[n * H1S + k]       = (u16)v;
      wt[(n + 1) * H1S + k] = (u16)(v >> 16);
    }
    __syncthreads();
    v8f acc0 = {}, acc1 = {};
#pragma unroll 4
    for (int kt = 0; kt < 16; ++kt) {
      v16bf a  = load_fragA(&h2[tm * 16 * H1S + kt * 32], H1S, lane);   // shared
      v16bf b0 = load_fragB(&wt[tn0 * 16 * H1S + kt * 32], H1S, lane);
      v16bf b1 = load_fragB(&wt[tn1 * 16 * H1S + kt * 32], H1S, lane);
      acc0 = wmma_bf16(a, b0, acc0);
      acc1 = wmma_bf16(a, b1, acc1);
    }
    const int gcol0 = nc * 64 + tn0 * 16 + col;
    const int gcol1 = nc * 64 + tn1 * 16 + col;
    float bv0 = ib3[gcol0];
    float bv1 = ib3[gcol1];
#pragma unroll
    for (int v = 0; v < 8; ++v) {
      int brow = rb * 64 + tm * 16 + v + 8 * half;
      float x0 = acc0[v] + bv0; x0 = x0 > 0.f ? x0 : 0.f;
      float x1 = acc1[v] + bv1; x1 = x1 > 0.f ? x1 : 0.f;
      unsafeAtomicAdd(&out[((size_t)i * 512 + brow) * 256 + gcol0], x0);
      unsafeAtomicAdd(&out[((size_t)i * 512 + brow) * 256 + gcol1], x1);
    }
    __syncthreads();
  }
}

extern "C" void kernel_launch(void* const* d_in, const int* in_sizes, int n_in,
                              void* d_out, int out_size, void* d_ws, size_t ws_size,
                              hipStream_t stream) {
  (void)in_sizes; (void)n_in; (void)out_size; (void)ws_size;
  const float* objs = (const float*)d_in[0];
  const float* tW1 = (const float*)d_in[1];  const float* tb1 = (const float*)d_in[2];
  const float* tW2 = (const float*)d_in[3];  const float* tb2 = (const float*)d_in[4];
  const float* tW3 = (const float*)d_in[5];  const float* tb3 = (const float*)d_in[6];
  const float* iW1 = (const float*)d_in[7];  const float* ib1 = (const float*)d_in[8];
  const float* iW2 = (const float*)d_in[9];  const float* ib2 = (const float*)d_in[10];
  const float* iW3 = (const float*)d_in[11]; const float* ib3 = (const float*)d_in[12];
  float* out = (float*)d_out;

  // workspace carve-up (all bf16 unless noted); total ~57 MB
  char* p = (char*)d_ws;
  u16* objs_bf = (u16*)p; p += (size_t)8192 * 256 * 2;
  u16* tW1b = (u16*)p;    p += (size_t)256 * 512 * 2;
  u16* tW2b = (u16*)p;    p += (size_t)512 * 512 * 2;
  u16* tW3b = (u16*)p;    p += (size_t)512 * 256 * 2;
  u16* iW1b = (u16*)p;    p += (size_t)512 * 512 * 2;
  u16* iW2b = (u16*)p;    p += (size_t)512 * 512 * 2;
  u16* iW3b = (u16*)p;    p += (size_t)512 * 256 * 2;
  u16* T1   = (u16*)p;    p += (size_t)8192 * 512 * 2;
  u16* T2   = (u16*)p;    p += (size_t)8192 * 512 * 2;
  float* P1 = (float*)p;  p += (size_t)8192 * 512 * 4;
  float* P2 = (float*)p;  p += (size_t)8192 * 512 * 4;

  auto cvt = [&](const float* s, u16* d, int n) {
    f32_to_bf16<<<(n / 2 + 255) / 256, 256, 0, stream>>>(s, d, n);
  };
  cvt(objs, objs_bf, 8192 * 256);
  cvt(tW1, tW1b, 256 * 512);  cvt(tW2, tW2b, 512 * 512);  cvt(tW3, tW3b, 512 * 256);
  cvt(iW1, iW1b, 512 * 512);  cvt(iW2, iW2b, 512 * 512);  cvt(iW3, iW3b, 512 * 256);

  dim3 blk(256);
  // transition MLP (T3 writes d_out = relu(.)+objs, so pair kernel just adds)
  gemm_bias_act<<<dim3(256, 8), blk, 0, stream>>>(objs_bf, tW1b, tb1, T1, nullptr, 8192, 256, 512, 0);
  gemm_bias_act<<<dim3(256, 8), blk, 0, stream>>>(T1, tW2b, tb2, T2, nullptr, 8192, 512, 512, 0);
  gemm_bias_act<<<dim3(256, 4), blk, 0, stream>>>(T2, tW3b, tb3, out, objs, 8192, 512, 256, 2);
  // interaction layer-1 factored: P1 = objs@iW1[:256], P2 = objs@iW1[256:]
  gemm_bias_act<<<dim3(256, 8), blk, 0, stream>>>(objs_bf, iW1b, nullptr, P1, nullptr, 8192, 256, 512, 1);
  gemm_bias_act<<<dim3(256, 8), blk, 0, stream>>>(objs_bf, iW1b + (size_t)256 * 512, nullptr, P2, nullptr, 8192, 256, 512, 1);

  // pair kernel: 16*16*8 blocks (240*8 active), ~195 KB dynamic LDS (<320 KB/WGP)
  size_t shmem = (size_t)3 * 64 * H1S * 2;
  (void)hipFuncSetAttribute((const void*)pair_kernel,
                            hipFuncAttributeMaxDynamicSharedMemorySize, (int)shmem);
  pair_kernel<<<dim3(8, 16, 16), blk, shmem, stream>>>(P1, P2, ib1, iW2b, ib2, iW3b, ib3, out);
}
